// NGPField_21165598835395
// MI455X (gfx1250) — matrix operand order repaired and checked by reading, not compiled, not run on hardware
//
#include <hip/hip_runtime.h>
#include <hip/hip_bf16.h>

typedef __attribute__((ext_vector_type(16))) _Float16 v16h;
typedef __attribute__((ext_vector_type(8)))  _Float16 v8h;
typedef __attribute__((ext_vector_type(8)))  float    v8f;

#define NPTS      524288
#define LOG2_T    19
#define TBL_MASK  ((1u << LOG2_T) - 1u)
#define NTILES    (NPTS / 16)
#define BLOCK     128
#define WPB       (BLOCK / 32)   // waves per block
#define GRID      1024

// floor(16 * (1024/16)^(l/15)) precomputed (matches reference SCALINGS)
__constant__ float c_scalings[16] = {
    16.f, 21.f, 27.f, 36.f, 48.f, 64.f, 84.f, 111.f,
    147.f, 194.f, 256.f, 337.f, 445.f, 588.f, 776.f, 1024.f};

// NeRF dir-encoding freqs: 2^linspace(0,4,4)
__constant__ float c_freqs[4] = {1.f, 2.5198420997897464f, 6.3496042078727974f, 16.f};

#define LDS_FENCE() asm volatile("s_wait_dscnt 0x0" ::: "memory")

static __device__ __forceinline__ v8f wmma16(v16h a, v16h b, v8f c) {
    return __builtin_amdgcn_wmma_f32_16x16x32_f16(
        /*neg_a=*/false, a, /*neg_b=*/false, b,
        /*c_mod=*/(short)0, c, /*reuse_a=*/false, /*reuse_b=*/false);
}

// Load one pre-swizzled B fragment (32 bytes contiguous per lane).
static __device__ __forceinline__ v16h load_B(const _Float16* frags, int frag, int lane) {
    return *(const v16h*)(frags + (frag * 32 + lane) * 16);
}

// Load a 16x32 A fragment from row-major f16 scratch with row stride 96.
static __device__ __forceinline__ v16h load_A96(const _Float16* buf, int lane, int kt) {
    const int m   = lane & 15;
    const int off = (lane < 16) ? 0 : 8;
    const _Float16* base = buf + m * 96 + kt + off;
    v8h lo = *(const v8h*)(base);
    v8h hi = *(const v8h*)(base + 16);
    v16h a;
#pragma unroll
    for (int i = 0; i < 8; ++i) { a[i] = lo[i]; a[i + 8] = hi[i]; }
    return a;
}

// Convert a [K x Nout] f32 weight into WMMA-B fragment layout in LDS (f16).
// Fragment index = ktIdx * (Nout/16) + ntIdx ; element i = frag*512 + lane*16 + e,
// maps to K = kt*32 + (lane<16?0:8) + (e&7) + (e>=8?16:0), N = nt*16 + (lane&15).
static __device__ void pack_B(const float* __restrict__ W, _Float16* dst,
                              int Kpad, int Nout, int Kreal) {
    const int nTiles = Nout >> 4;
    const int total  = (Kpad >> 5) * nTiles * 512;
    for (int i = threadIdx.x; i < total; i += blockDim.x) {
        const int frag = i >> 9;
        const int r    = i & 511;
        const int ln   = r >> 4;
        const int e    = r & 15;
        const int kt   = frag / nTiles;
        const int nt   = frag - kt * nTiles;
        const int k    = (kt << 5) + ((ln < 16) ? 0 : 8) + (e & 7) + ((e & 8) << 1);
        const int n    = (nt << 4) + (ln & 15);
        dst[i] = (k < Kreal) ? (_Float16)W[k * Nout + n] : (_Float16)0.f;
    }
}

__global__ __launch_bounds__(BLOCK)
void ngp_field_kernel(const float* __restrict__ pos, const float* __restrict__ dir,
                      const float* __restrict__ table,
                      const float* __restrict__ w0, const float* __restrict__ b0,
                      const float* __restrict__ w1, const float* __restrict__ b1,
                      const float* __restrict__ w2, const float* __restrict__ b2,
                      const float* __restrict__ wr, const float* __restrict__ br,
                      const float* __restrict__ wc, const float* __restrict__ bc,
                      const float* __restrict__ wd, const float* __restrict__ bd,
                      float* __restrict__ out) {
    // Weights pre-swizzled into B-fragment layout (f16)
    __shared__ __align__(32) _Float16 s_w0f[4 * 512];   // 32x64 -> 4 frags
    __shared__ __align__(32) _Float16 s_w1f[8 * 512];   // 64x64 -> 8 frags
    __shared__ __align__(32) _Float16 s_w2f[8 * 512];   // 64x64 -> 8 frags
    __shared__ __align__(32) _Float16 s_wrf[6 * 512];   // 96(91)x32 -> 6 frags
    // Per-wave activation scratch: two ping-pong [16 x 96] f16 buffers
    __shared__ __align__(32) _Float16 s_act[WPB][2][16 * 96];
    __shared__ float s_b0[64], s_b1[64], s_b2[64], s_wd[64];
    __shared__ float s_br[32], s_wc[96], s_bc[3], s_bd[1];

    // ---- cooperative weight preload / f16 swizzle ----
    pack_B(w0, s_w0f, 32, 64, 32);
    pack_B(w1, s_w1f, 64, 64, 64);
    pack_B(w2, s_w2f, 64, 64, 64);
    pack_B(wr, s_wrf, 96, 32, 91);   // rows 91..95 zero-padded
    for (int i = threadIdx.x; i < 64; i += blockDim.x) {
        s_b0[i] = b0[i]; s_b1[i] = b1[i]; s_b2[i] = b2[i]; s_wd[i] = wd[i];
    }
    for (int i = threadIdx.x; i < 32; i += blockDim.x) s_br[i] = br[i];
    for (int i = threadIdx.x; i < 96; i += blockDim.x) s_wc[i] = wc[i];
    if (threadIdx.x == 0) { s_bc[0] = bc[0]; s_bc[1] = bc[1]; s_bc[2] = bc[2]; s_bd[0] = bd[0]; }
    __syncthreads();

    const int lane   = threadIdx.x & 31;
    const int w      = threadIdx.x >> 5;
    const int hi     = lane >> 4;
    const int nWaves = GRID * WPB;
    const int wave0  = blockIdx.x * WPB + w;
    _Float16* bufA = &s_act[w][0][0];
    _Float16* bufB = &s_act[w][1][0];

    for (int tile = wave0; tile < NTILES; tile += nWaves) {
        const int p = tile * 16 + (lane & 15);

        // Prefetch next tile's streaming inputs into cache hierarchy
        const int tn = tile + nWaves;
        if (tn < NTILES) {
            __builtin_prefetch(&pos[3 * (tn * 16 + (lane & 15))], 0, 3);
            __builtin_prefetch(&dir[3 * (tn * 16 + (lane & 15))], 0, 3);
        }

        // =================== hash encoding, directly in A-layout ==========
        // Lane (l)  : point = l%16.  K-slots of this lane = levels
        //   l<16 : levels 0..3 (e0..7), levels  8..11 (e8..15)
        //   l>=16: levels 4..7 (e0..7), levels 12..15 (e8..15)
        const float px = pos[3 * p + 0];
        const float py = pos[3 * p + 1];
        const float pz = pos[3 * p + 2];
        const int lvl0 = (lane < 16) ? 0 : 4;
        v16h aenc;
#pragma unroll
        for (int g = 0; g < 2; ++g) {
#pragma unroll
            for (int j = 0; j < 4; ++j) {
                const int lvl = lvl0 + g * 8 + j;
                const float s = c_scalings[lvl];
                const float sx = px * s, sy = py * s, sz = pz * s;
                const float fx = floorf(sx), fy = floorf(sy), fz = floorf(sz);
                const float ox = sx - fx, oy = sy - fy, oz = sz - fz;
                const uint32_t ix = (uint32_t)(int)fx;
                const uint32_t iy = (uint32_t)(int)fy;
                const uint32_t iz = (uint32_t)(int)fz;
                const uint32_t lvl_off = (uint32_t)lvl << LOG2_T;

                // Per-axis hash terms & weights (hoisted out of the corner loop)
                const uint32_t hx[2] = {ix, ix + 1u};
                const uint32_t hy[2] = {iy * 2654435761u, (iy + 1u) * 2654435761u};
                const uint32_t hz[2] = {iz * 805459861u, (iz + 1u) * 805459861u};
                const float wxs[2] = {1.f - ox, ox};
                const float wys[2] = {1.f - oy, oy};
                const float wzs[2] = {1.f - oz, oz};

                // Phase 1: all 8 indices + weights (no memory ops)
                uint32_t idxs[8];
                float    wgts[8];
#pragma unroll
                for (int c = 0; c < 8; ++c) {
                    const int bx = c & 1, by = (c >> 1) & 1, bz = (c >> 2) & 1;
                    idxs[c] = ((hx[bx] ^ hy[by] ^ hz[bz]) & TBL_MASK) + lvl_off;
                    wgts[c] = wxs[bx] * wys[by] * wzs[bz];
                }
                // Phase 2: 8 independent gathers -> deep load clause, one wait
                float2 tv[8];
#pragma unroll
                for (int c = 0; c < 8; ++c) tv[c] = ((const float2*)table)[idxs[c]];
                // Phase 3: short FMA tree (2 feats)
                float f0 = (fmaf(wgts[1], tv[1].x, wgts[0] * tv[0].x) +
                            fmaf(wgts[3], tv[3].x, wgts[2] * tv[2].x)) +
                           (fmaf(wgts[5], tv[5].x, wgts[4] * tv[4].x) +
                            fmaf(wgts[7], tv[7].x, wgts[6] * tv[6].x));
                float f1 = (fmaf(wgts[1], tv[1].y, wgts[0] * tv[0].y) +
                            fmaf(wgts[3], tv[3].y, wgts[2] * tv[2].y)) +
                           (fmaf(wgts[5], tv[5].y, wgts[4] * tv[4].y) +
                            fmaf(wgts[7], tv[7].y, wgts[6] * tv[6].y));
                aenc[g * 8 + j * 2 + 0] = (_Float16)f0;
                aenc[g * 8 + j * 2 + 1] = (_Float16)f1;
            }
        }

        // =================== layer 0: [16x32] @ [32x64], ReLU =============
        // Batched: all B frags -> all WMMAs back-to-back -> all epilogues
        {
            v16h bf[4];
#pragma unroll
            for (int nt = 0; nt < 4; ++nt) bf[nt] = load_B(s_w0f, nt, lane);
            v8f c[4];
#pragma unroll
            for (int nt = 0; nt < 4; ++nt) { v8f z = {}; c[nt] = wmma16(aenc, bf[nt], z); }
#pragma unroll
            for (int nt = 0; nt < 4; ++nt) {
                const float bias = s_b0[nt * 16 + (lane & 15)];
#pragma unroll
                for (int v = 0; v < 8; ++v) {
                    const float x = c[nt][v] + bias;
                    bufA[(v + 8 * hi) * 96 + nt * 16 + (lane & 15)] = (_Float16)fmaxf(x, 0.f);
                }
            }
        }

        // =================== layer 1: [16x64] @ [64x64], ReLU =============
        LDS_FENCE();
        {
            v16h a0 = load_A96(bufA, lane, 0);
            v16h a1 = load_A96(bufA, lane, 32);
            LDS_FENCE();
            // bufA free now: write dir-encoding (cols 0..26) and zero pad (91..95)
            if (lane < 16) {
                const int m = lane;
                const float d3[3] = {dir[3 * p], dir[3 * p + 1], dir[3 * p + 2]};
                const float TWO_PI = 6.28318530717958647692f;
                int col = 0;
#pragma unroll
                for (int dd = 0; dd < 3; ++dd)
#pragma unroll
                    for (int k = 0; k < 4; ++k) {
                        const float v = TWO_PI * d3[dd] * c_freqs[k];
                        bufA[m * 96 + col]      = (_Float16)__sinf(v);
                        bufA[m * 96 + 12 + col] = (_Float16)__cosf(v);
                        ++col;
                    }
                bufA[m * 96 + 24] = (_Float16)d3[0];
                bufA[m * 96 + 25] = (_Float16)d3[1];
                bufA[m * 96 + 26] = (_Float16)d3[2];
#pragma unroll
                for (int cc = 91; cc < 96; ++cc) bufA[m * 96 + cc] = (_Float16)0.f;
            }
            v16h bf[8];
#pragma unroll
            for (int i = 0; i < 8; ++i) bf[i] = load_B(s_w1f, i, lane);
            v8f c[4];
#pragma unroll
            for (int nt = 0; nt < 4; ++nt) { v8f z = {}; c[nt] = wmma16(a0, bf[nt], z); }
#pragma unroll
            for (int nt = 0; nt < 4; ++nt) c[nt] = wmma16(a1, bf[4 + nt], c[nt]);
#pragma unroll
            for (int nt = 0; nt < 4; ++nt) {
                const float bias = s_b1[nt * 16 + (lane & 15)];
#pragma unroll
                for (int v = 0; v < 8; ++v) {
                    const float x = c[nt][v] + bias;
                    bufB[(v + 8 * hi) * 96 + nt * 16 + (lane & 15)] = (_Float16)fmaxf(x, 0.f);
                }
            }
        }

        // =================== layer 2: [16x64] @ [64x64] (linear) ==========
        LDS_FENCE();
        {
            v16h a0 = load_A96(bufB, lane, 0);
            v16h a1 = load_A96(bufB, lane, 32);
            v16h bf[8];
#pragma unroll
            for (int i = 0; i < 8; ++i) bf[i] = load_B(s_w2f, i, lane);
            LDS_FENCE();
            v8f c[4];
#pragma unroll
            for (int nt = 0; nt < 4; ++nt) { v8f z = {}; c[nt] = wmma16(a0, bf[nt], z); }
#pragma unroll
            for (int nt = 0; nt < 4; ++nt) c[nt] = wmma16(a1, bf[4 + nt], c[nt]);
#pragma unroll
            for (int nt = 0; nt < 4; ++nt) {
                const float bias = s_b2[nt * 16 + (lane & 15)];
#pragma unroll
                for (int v = 0; v < 8; ++v)   // base -> concat cols 27..90
                    bufA[(v + 8 * hi) * 96 + 27 + nt * 16 + (lane & 15)] = (_Float16)(c[nt][v] + bias);
            }
        }

        // =================== rgb layer: [16x96] @ [96x32] =================
        LDS_FENCE();
        float* rgbh = (float*)bufB;   // bufB (act1) is dead; reuse as [16x32] f32
        {
            v16h a0 = load_A96(bufA, lane, 0);
            v16h a1 = load_A96(bufA, lane, 32);
            v16h a2 = load_A96(bufA, lane, 64);
            v16h bf[6];
#pragma unroll
            for (int i = 0; i < 6; ++i) bf[i] = load_B(s_wrf, i, lane);
            v8f c[2];
#pragma unroll
            for (int nt = 0; nt < 2; ++nt) { v8f z = {}; c[nt] = wmma16(a0, bf[nt], z); }
#pragma unroll
            for (int nt = 0; nt < 2; ++nt) c[nt] = wmma16(a1, bf[2 + nt], c[nt]);
#pragma unroll
            for (int nt = 0; nt < 2; ++nt) c[nt] = wmma16(a2, bf[4 + nt], c[nt]);
#pragma unroll
            for (int nt = 0; nt < 2; ++nt) {
                const float bias = s_br[nt * 16 + (lane & 15)];
#pragma unroll
                for (int v = 0; v < 8; ++v)
                    rgbh[(v + 8 * hi) * 32 + nt * 16 + (lane & 15)] = c[nt][v] + bias;
            }
        }
        LDS_FENCE();

        // =================== heads: density + rgb (skinny dots) ===========
        if (lane < 16) {
            const int m = lane;
            float acc = s_bd[0];
#pragma unroll
            for (int n = 0; n < 64; ++n)
                acc = fmaf((float)bufA[m * 96 + 27 + n], s_wd[n], acc);
            const float dens = (acc > 20.f) ? acc : __logf(1.f + __expf(acc));
            float r[3];
#pragma unroll
            for (int ch = 0; ch < 3; ++ch) {
                float a = s_bc[ch];
#pragma unroll
                for (int n = 0; n < 32; ++n)
                    a = fmaf(rgbh[m * 32 + n], s_wc[n * 3 + ch], a);
                r[ch] = 1.f / (1.f + __expf(-a));
            }
            *(float4*)&out[4 * p] = make_float4(r[0], r[1], r[2], dens);
        }
        LDS_FENCE();
    }
}

extern "C" void kernel_launch(void* const* d_in, const int* in_sizes, int n_in,
                              void* d_out, int out_size, void* d_ws, size_t ws_size,
                              hipStream_t stream) {
    (void)in_sizes; (void)n_in; (void)out_size; (void)d_ws; (void)ws_size;
    const float* pos   = (const float*)d_in[0];
    const float* dirs  = (const float*)d_in[1];
    const float* table = (const float*)d_in[2];
    const float* w0 = (const float*)d_in[3];  const float* b0 = (const float*)d_in[4];
    const float* w1 = (const float*)d_in[5];  const float* b1 = (const float*)d_in[6];
    const float* w2 = (const float*)d_in[7];  const float* b2 = (const float*)d_in[8];
    const float* wr = (const float*)d_in[9];  const float* br = (const float*)d_in[10];
    const float* wc = (const float*)d_in[11]; const float* bc = (const float*)d_in[12];
    const float* wd = (const float*)d_in[13]; const float* bd = (const float*)d_in[14];
    ngp_field_kernel<<<GRID, BLOCK, 0, stream>>>(
        pos, dirs, table, w0, b0, w1, b1, w2, b2, wr, br, wc, bc, wd, bd,
        (float*)d_out);
}
